// SLinear_72138270703709
// MI455X (gfx1250) — matrix operand order repaired and checked by reading, not compiled, not run on hardware
//
#include <hip/hip_runtime.h>

// SLinear (per-position batched GEMM) for gfx1250 / MI455X.
// out[b,p,o] = sum_i x[b,p,i] * w[p,i,o] + bias[p,o]
// B=8192, P=15, I=O=512, fp32 in/out.
//
// Strategy: bf16x3 split-precision WMMA (xh*wh + xh*wl + xl*wh) with f32
// accumulation via v_wmma_f32_16x16x32_bf16. 128x128 output tile per block,
// K staged through LDS in 32-wide chunks as hi/lo bf16.

typedef __bf16 bf16;
typedef __attribute__((ext_vector_type(8)))  bf16  bf16x8;
typedef __attribute__((ext_vector_type(16))) bf16  v16bf;
typedef __attribute__((ext_vector_type(8)))  float v8f;
typedef __attribute__((ext_vector_type(4)))  float f32x4;

namespace {
constexpr int PEP    = 15;
constexpr int CIN    = 512;
constexpr int COUT   = 512;
constexpr int TM     = 128;
constexpr int TN     = 128;
constexpr int KC     = 32;          // k-chunk per LDS stage (= WMMA bf16 K)
constexpr int LDK    = 40;          // padded LDS row stride (bf16): 80B, 16B-aligned rows
constexpr int KTILES = CIN / KC;    // 16
}

__device__ __forceinline__ void split_f32_bf16(float x, bf16 &h, bf16 &l) {
  h = (bf16)x;                 // RNE truncation to bf16 (top 8 mantissa bits)
  l = (bf16)(x - (float)h);    // residual (~next 8 mantissa bits)
}

__device__ __forceinline__ v16bf frag_from_halves(const bf16 *p0, const bf16 *p1) {
  bf16x8 h0 = *reinterpret_cast<const bf16x8 *>(p0);   // ds_load_b128
  bf16x8 h1 = *reinterpret_cast<const bf16x8 *>(p1);   // ds_load_b128
  v16bf r;
#pragma unroll
  for (int i = 0; i < 8; ++i) { r[i] = h0[i]; r[i + 8] = h1[i]; }
  return r;
}

__global__ __launch_bounds__(256, 2) void slinear_bf16x3_wmma(
    const float *__restrict__ X, const float *__restrict__ W,
    const float *__restrict__ Bias, float *__restrict__ Out)
{
  // A tiles row-major [m][k]; B tiles transposed [n][k] so each lane's
  // two 8-K fragment halves are contiguous. Stride 40 (80B) avoids bank conflicts
  // while keeping every 8-element half 16B-aligned.
  __shared__ __align__(16) bf16 Ah[TM][LDK];
  __shared__ __align__(16) bf16 Al[TM][LDK];
  __shared__ __align__(16) bf16 Bh[TN][LDK];
  __shared__ __align__(16) bf16 Bl[TN][LDK];

  const int bid = blockIdx.x;
  const int mt  = bid & 63;         // 64 M-tiles of 128
  const int nt  = (bid >> 6) & 3;   // 4  N-tiles of 128
  const int p   = bid >> 8;         // 15 positions

  const int tid  = threadIdx.x;
  const int lane = tid & 31;
  const int wave = tid >> 5;        // 8 waves
  const int wm   = wave & 1;        // 2 wave rows -> 64 M rows each
  const int wn   = wave >> 1;       // 4 wave cols -> 32 N cols each

  const int l16   = lane & 15;
  const int lhalf = lane >> 4;      // 0: lanes 0-15, 1: lanes 16-31

  v8f acc[4][2] = {};

  // staging coordinates
  const int arow0 = tid >> 3;          // 0..31 (+32/pass): A row
  const int acol4 = (tid & 7) << 2;    // 0..28: A k offset (float4)
  const int bk0   = tid >> 5;          // 0..7 (+8/pass): B k row
  const int bn4   = (tid & 31) << 2;   // 0..124: B n offset (float4)

  const size_t xRowStride = (size_t)PEP * CIN;
  const float *Xblk = X + ((size_t)(mt * TM) * PEP + p) * CIN;
  const float *Wblk = W + (size_t)p * CIN * COUT + (size_t)nt * TN;

  for (int kc = 0; kc < KTILES; ++kc) {
    const int k0 = kc * KC;

    // ---- stage A tile: 128x32 f32 -> hi/lo bf16 ----
#pragma unroll
    for (int pass = 0; pass < 4; ++pass) {
      const int r = arow0 + pass * 32;
      f32x4 v = *reinterpret_cast<const f32x4 *>(
          Xblk + (size_t)r * xRowStride + k0 + acol4);
#pragma unroll
      for (int j = 0; j < 4; ++j) {
        bf16 h, l;
        split_f32_bf16(v[j], h, l);
        Ah[r][acol4 + j] = h;
        Al[r][acol4 + j] = l;
      }
    }

    // ---- stage B tile: 32x128 f32 -> hi/lo bf16, transposed to [n][k] ----
#pragma unroll
    for (int pass = 0; pass < 4; ++pass) {
      const int kr = bk0 + pass * 8;
      f32x4 v = *reinterpret_cast<const f32x4 *>(
          Wblk + (size_t)(k0 + kr) * COUT + bn4);
#pragma unroll
      for (int j = 0; j < 4; ++j) {
        bf16 h, l;
        split_f32_bf16(v[j], h, l);
        Bh[bn4 + j][kr] = h;
        Bl[bn4 + j][kr] = l;
      }
    }

    // prefetch next k-chunk into cache (global_prefetch_b8)
    if (kc + 1 < KTILES) {
      __builtin_prefetch(Xblk + (size_t)arow0 * xRowStride + k0 + KC + acol4, 0, 1);
      __builtin_prefetch(Wblk + (size_t)(k0 + KC + bk0) * COUT + bn4, 0, 1);
    }

    __syncthreads();

    // ---- 3-term split WMMA: x*w ~= xh*wh + xh*wl + xl*wh ----
#pragma unroll
    for (int ni = 0; ni < 2; ++ni) {
      // B fragment: lane holds column N = n, K in {bko..bko+7, bko+16..bko+23}
      const int n   = wn * 32 + ni * 16 + l16;
      const int bko = lhalf * 8;
      v16bf bh = frag_from_halves(&Bh[n][bko], &Bh[n][bko + 16]);
      v16bf bl = frag_from_halves(&Bl[n][bko], &Bl[n][bko + 16]);
#pragma unroll
      for (int mi = 0; mi < 4; ++mi) {
        // A fragment: lane holds row M = m, K in {ako..ako+7, ako+16..ako+23}
        const int m   = wm * 64 + mi * 16 + l16;
        const int ako = lhalf * 8;
        v16bf ahf = frag_from_halves(&Ah[m][ako], &Ah[m][ako + 16]);
        v16bf alf = frag_from_halves(&Al[m][ako], &Al[m][ako + 16]);
        acc[mi][ni] = __builtin_amdgcn_wmma_f32_16x16x32_bf16(
            false, ahf, false, bh, (short)0, acc[mi][ni], false, false);
        acc[mi][ni] = __builtin_amdgcn_wmma_f32_16x16x32_bf16(
            false, ahf, false, bl, (short)0, acc[mi][ni], false, false);
        acc[mi][ni] = __builtin_amdgcn_wmma_f32_16x16x32_bf16(
            false, alf, false, bh, (short)0, acc[mi][ni], false, false);
      }
    }

    __syncthreads();
  }

  // ---- epilogue: bias add + coalesced f32 stores ----
  // C/D layout: lane (0-15 | 16-31) -> N = l16; VGPR r -> M = r (+8 for hi half)
#pragma unroll
  for (int ni = 0; ni < 2; ++ni) {
    const int n = nt * TN + wn * 32 + ni * 16 + l16;
    const float bv = Bias[p * COUT + n];
#pragma unroll
    for (int mi = 0; mi < 4; ++mi) {
#pragma unroll
      for (int r = 0; r < 8; ++r) {
        const int m = mt * TM + wm * 64 + mi * 16 + (lhalf << 3) + r;
        Out[((size_t)m * PEP + p) * COUT + n] = acc[mi][ni][r] + bv;
      }
    }
  }
}

extern "C" void kernel_launch(void* const* d_in, const int* in_sizes, int n_in,
                              void* d_out, int out_size, void* d_ws, size_t ws_size,
                              hipStream_t stream) {
  (void)in_sizes; (void)n_in; (void)out_size; (void)d_ws; (void)ws_size;
  const float* X    = (const float*)d_in[0];  // [8192,15,512]
  const float* W    = (const float*)d_in[1];  // [15,512,512]
  const float* Bias = (const float*)d_in[2];  // [15,512]
  float* Out        = (float*)d_out;          // [8192,15,512]

  const int grid = (8192 / 128) * (512 / 128) * 15;  // 3840 blocks
  slinear_bf16x3_wmma<<<grid, 256, 0, stream>>>(X, W, Bias, Out);
}
